// WyckoffGNN_84610855731761
// MI455X (gfx1250) — compile-verified, matrix operand order checked by reading, not run on hardware
//
#include <hip/hip_runtime.h>

typedef __attribute__((ext_vector_type(16))) _Float16 v16h;
typedef __attribute__((ext_vector_type(8)))  float    v8f;
typedef __attribute__((ext_vector_type(4)))  int      v4i;

// Async global->LDS staging (CDNA5 GLOBAL_LOAD_ASYNC_TO_LDS_B128, ASYNCcnt),
// guarded so the file still compiles if the builtin is not declared.
#if defined(__HIP_DEVICE_COMPILE__) && __has_builtin(__builtin_amdgcn_global_load_async_to_lds_b128)
#define USE_ASYNC_LDS 1
#else
#define USE_ASYNC_LDS 0
#endif

#if USE_ASYNC_LDS
// Copy 16 bytes global -> LDS via the async DMA path (tracked by ASYNCcnt).
// Builtin signature (from clang diagnostic): (v4i AS1*, v4i AS3*, imm off, imm cpol).
__device__ __forceinline__ void async_copy_b128(const _Float16* gsrc, _Float16* ldst) {
    __builtin_amdgcn_global_load_async_to_lds_b128(
        (__attribute__((address_space(1))) v4i*)(v4i*)gsrc,
        (__attribute__((address_space(3))) v4i*)(v4i*)ldst,
        0, 0);
}
#endif

__device__ __forceinline__ void wait_async_lds() {
#if defined(__HIP_DEVICE_COMPILE__)
#if __has_builtin(__builtin_amdgcn_s_wait_asynccnt)
    __builtin_amdgcn_s_wait_asynccnt(0);
#else
    asm volatile("s_wait_asynccnt 0x0" ::: "memory");
#endif
#endif
}

__device__ __forceinline__ float silu_f(float x) {
    return x / (1.0f + __expf(-x));
}

// Load one 16x32 (f16) WMMA operand fragment from an LDS row.
// rowptr points at the 32-half K-block for this lane's row (M for A, N for B).
// Per CDNA5 ISA layout: lanes 0-15 (g=0) hold K = {0..7, 16..23},
// lanes 16-31 (g=1) hold K = {8..15, 24..31}; vector elem t -> VGPR t/2 half t%2.
// As dwords: indices {o..o+3, o+8..o+11} with o = 4*g.
__device__ __forceinline__ v16h load_frag(const _Float16* rowptr, int g) {
    const unsigned int* p = (const unsigned int*)rowptr;
    union { v16h v; unsigned int u[8]; } r;
    const int o = g * 4;
    r.u[0] = p[o + 0];  r.u[1] = p[o + 1];
    r.u[2] = p[o + 2];  r.u[3] = p[o + 3];
    r.u[4] = p[o + 8];  r.u[5] = p[o + 9];
    r.u[6] = p[o + 10]; r.u[7] = p[o + 11];
    return r.v;
}

// ---------------------------------------------------------------------------
// Generic WMMA GEMM:  C[M,N] = act(A[M,K] @ W[K,N] + bias)
// A: f16 row-major. W: f16 row-major [K][N]. Outputs: optional f32 and/or f16.
// M % 64 == 0, K % 32 == 0 (always true here). N arbitrary (bounds-guarded).
// Block: 128 threads = 4 waves; each wave owns 16 rows, block owns 64x64 tile.
// A tile staged via async global->LDS B128 copies when available.
// ---------------------------------------------------------------------------
__global__ __launch_bounds__(128) void gemm_wmma_kernel(
    const _Float16* __restrict__ A, const _Float16* __restrict__ W,
    const float* __restrict__ bias, float* __restrict__ Cf,
    _Float16* __restrict__ Ch, int M, int K, int N, int doSilu)
{
    // 80B row stride: rows stay 16B aligned for B128 LDS writes and the
    // 20-dword bank stride keeps 16-row fragment reads conflict-free.
    __shared__ __align__(16) _Float16 As[64][40];
    __shared__ _Float16 Ws[64][34];   // stored transposed: Ws[n][k], 17-bank stride

    const int tid  = threadIdx.x;
    const int wave = tid >> 5;
    const int lane = tid & 31;
    const int g    = lane >> 4;
    const int ln   = lane & 15;
    const int m0b  = blockIdx.y * 64;
    const int n0b  = blockIdx.x * 64;

    v8f acc[4];
    {
        v8f zer = {0.f, 0.f, 0.f, 0.f, 0.f, 0.f, 0.f, 0.f};
        for (int j = 0; j < 4; ++j) acc[j] = zer;
    }

    for (int k0 = 0; k0 < K; k0 += 32) {
#if USE_ASYNC_LDS
        // Stage A tile 64x32 halfs as 4 x B128 chunks per row, async into LDS.
        for (int i = tid; i < 64 * 4; i += 128) {
            int r = i >> 2;
            int c = i & 3;
            async_copy_b128(A + (size_t)(m0b + r) * K + k0 + c * 8, &As[r][c * 8]);
        }
#else
        for (int i = tid; i < 64 * 16; i += 128) {
            int r  = i >> 4;
            int kk = i & 15;
            unsigned int v =
                *(const unsigned int*)(A + (size_t)(m0b + r) * K + k0 + kk * 2);
            *(unsigned int*)&As[r][kk * 2] = v;
        }
#endif
        // Stage W tile 32x64, transposed into Ws[n][k] (coalesced along N)
        for (int i = tid; i < 32 * 64; i += 128) {
            int k  = i >> 6;
            int n  = i & 63;
            int gn = n0b + n;
            _Float16 v = (_Float16)0.f;
            if (gn < N) v = W[(size_t)(k0 + k) * N + gn];
            Ws[n][k] = v;
        }
#if USE_ASYNC_LDS
        wait_async_lds();
#endif
        __syncthreads();

        // Issue all fragment loads first, then the WMMAs, so the scheduler can
        // overlap DS traffic with matrix issue using partial dscnt waits.
        v16h af = load_frag(&As[16 * wave + ln][0], g);
        v16h bf[4];
        for (int j = 0; j < 4; ++j) bf[j] = load_frag(&Ws[16 * j + ln][0], g);
        for (int j = 0; j < 4; ++j) {
            acc[j] = __builtin_amdgcn_wmma_f32_16x16x32_f16(
                false, af, false, bf[j], (short)0, acc[j], false, false);
        }
        __syncthreads();
    }

    // Epilogue: D elem (r, lane): m = 16*wave + r + 8*(lane/16), n = 16*j + ln
    const int mbase = m0b + 16 * wave + 8 * g;
    for (int j = 0; j < 4; ++j) {
        int n = n0b + 16 * j + ln;
        if (n < N) {
            float bv = bias ? bias[n] : 0.0f;
            for (int r = 0; r < 8; ++r) {
                float v = acc[j][r] + bv;
                if (doSilu) v = silu_f(v);
                size_t off = (size_t)(mbase + r) * N + n;
                if (Cf) Cf[off] = v;
                if (Ch) Ch[off] = (_Float16)v;
            }
        }
    }
}

// ---------------------------------------------------------------------------
// Edge kernel: per 64-edge tile, compute
//   x   = silu(U[dst] + V[src])           (b_a0 folded into U)
//   y   = silu(x @ W_a1 + b_a1)           (64x384 @ 384x384, WMMA)
//   a   = y @ w_a2 + b_a2                 (scalar per edge)
//   agg[dst] += a * psi[src]              (global f32 atomics)
// 256 threads = 8 waves: wave (wr = w%4) -> 16 edges, (wc = w/4) -> 192 cols.
// ---------------------------------------------------------------------------
__global__ __launch_bounds__(256) void edge_kernel(
    const int* __restrict__ ei, const float* __restrict__ U,
    const float* __restrict__ V, const float* __restrict__ b1,
    const _Float16* __restrict__ W1, const float* __restrict__ w2,
    const float* __restrict__ b2, const float* __restrict__ psi,
    float* __restrict__ agg)
{
    __shared__ _Float16 xs[64][386];   // 64 edges x 384 + pad
    __shared__ _Float16 wt[384][34];   // W1 K-slice transposed [n][k]
    __shared__ float    b1s[384];
    __shared__ float    w2s[384];
    __shared__ float    a_s[64];
    __shared__ int      srcs[64];
    __shared__ int      dsts[64];

    const int tid = threadIdx.x;
    const int e0  = blockIdx.x * 64;

    if (tid < 64) {
        srcs[tid] = ei[e0 + tid];           // edge_index[0] = src
        dsts[tid] = ei[65536 + e0 + tid];   // edge_index[1] = dst
        a_s[tid]  = 0.0f;
    }
    for (int i = tid; i < 384; i += 256) {
        b1s[i] = b1[i];
        w2s[i] = w2[i];
    }
    __syncthreads();

    // Build x in LDS (f16)
    for (int i = tid; i < 64 * 384; i += 256) {
        int e = i / 384;
        int k = i - e * 384;
        float xv = U[(size_t)dsts[e] * 384 + k] + V[(size_t)srcs[e] * 384 + k];
        xs[e][k] = (_Float16)silu_f(xv);
    }
    __syncthreads();

    const int wave = tid >> 5;
    const int lane = tid & 31;
    const int g    = lane >> 4;
    const int ln   = lane & 15;
    const int wr   = wave & 3;
    const int wc   = wave >> 2;

    v8f acc[12];
    {
        v8f zer = {0.f, 0.f, 0.f, 0.f, 0.f, 0.f, 0.f, 0.f};
        for (int j = 0; j < 12; ++j) acc[j] = zer;
    }

    for (int k0 = 0; k0 < 384; k0 += 32) {
        // Stage W1[k0:k0+32][0:384] transposed (dword global loads along N)
        for (int i = tid; i < 32 * 192; i += 256) {
            int k  = i / 192;
            int n2 = i - k * 192;
            union { unsigned int u; _Float16 h[2]; } w;
            w.u = *(const unsigned int*)(W1 + (size_t)(k0 + k) * 384 + n2 * 2);
            wt[n2 * 2][k]     = w.h[0];
            wt[n2 * 2 + 1][k] = w.h[1];
        }
        __syncthreads();

        // Issue A + all 12 B fragment loads, then the WMMA chain.
        v16h af = load_frag(&xs[16 * wr + ln][k0], g);
        v16h bf[12];
        for (int j = 0; j < 12; ++j)
            bf[j] = load_frag(&wt[192 * wc + 16 * j + ln][0], g);
        for (int j = 0; j < 12; ++j) {
            acc[j] = __builtin_amdgcn_wmma_f32_16x16x32_f16(
                false, af, false, bf[j], (short)0, acc[j], false, false);
        }
        __syncthreads();
    }

    // a[e] = sum_n silu(y[e][n]) * w2[n]  (partial per lane, LDS atomic reduce)
    float part[8];
    for (int r = 0; r < 8; ++r) part[r] = 0.0f;
    for (int j = 0; j < 12; ++j) {
        int   n  = 192 * wc + 16 * j + ln;
        float bv = b1s[n];
        float wv = w2s[n];
        for (int r = 0; r < 8; ++r) {
            float y = silu_f(acc[j][r] + bv);
            part[r] += y * wv;
        }
    }
    for (int r = 0; r < 8; ++r) {
        atomicAdd(&a_s[16 * wr + 8 * g + r], part[r]);
    }
    __syncthreads();
    if (tid < 64) a_s[tid] += b2[0];
    __syncthreads();

    // Scatter: agg[dst] += a * psi[src]
    for (int i = tid; i < 64 * 128; i += 256) {
        int e = i >> 7;
        int c = i & 127;
        float val = a_s[e] * psi[(size_t)srcs[e] * 128 + c];
        atomicAdd(agg + (size_t)dsts[e] * 128 + c, val);
    }
}

// ---------------------------------------------------------------------------
// Node embedding: h[n] = zero_dof_emb[x0[n]] (n<4096)
//                 h[n] = inf_dof_emb[x_inf[n-4096]] @ inf_W + inf_b  otherwise
// One block (128 threads) per node.
// ---------------------------------------------------------------------------
__global__ __launch_bounds__(128) void emb_h_kernel(
    const float* __restrict__ zero_emb, const float* __restrict__ inf_emb,
    const float* __restrict__ infW, const float* __restrict__ infb,
    const int* __restrict__ x0, const int* __restrict__ xinf,
    float* __restrict__ h)
{
    __shared__ float ev[100];
    const int n = blockIdx.x;
    const int c = threadIdx.x;
    if (n < 4096) {
        h[(size_t)n * 128 + c] = zero_emb[(size_t)x0[n] * 128 + c];
    } else {
        const int r = n - 4096;
        if (c < 100) ev[c] = inf_emb[xinf[(size_t)r * 100 + c]];
        __syncthreads();
        float s = infb[c];
        for (int k = 0; k < 100; ++k) s += ev[k] * infW[k * 128 + c];
        h[(size_t)n * 128 + c] = s;
    }
}

__global__ void hdof_kernel(
    const float* __restrict__ dof_emb, const float* __restrict__ pos_emb,
    const float* __restrict__ sg_emb, const float* __restrict__ time_emb,
    const int* __restrict__ wy, const int* __restrict__ sg,
    const int* __restrict__ tt, float* __restrict__ hdof)
{
    int idx = blockIdx.x * blockDim.x + threadIdx.x;   // 8192*64 exact
    int n = idx >> 6, c = idx & 63;
    int gidx  = n >> 8;                                // 256 nodes per graph
    int dflag = (n < 4096) ? 1 : 0;
    hdof[idx] = dof_emb[dflag * 64 + c] + pos_emb[wy[n] * 64 + c]
              + sg_emb[sg[gidx] * 64 + c] + time_emb[tt[gidx] * 64 + c];
}

__global__ void buildz_kernel(const float* __restrict__ h,
                              const float* __restrict__ hdof,
                              _Float16* __restrict__ z,
                              float* __restrict__ agg)
{
    int idx = blockIdx.x * blockDim.x + threadIdx.x;   // 8192*192 exact
    int n = idx / 192;
    int c = idx - n * 192;
    float v = (c < 128) ? h[n * 128 + c] : hdof[n * 64 + (c - 128)];
    z[idx] = (_Float16)v;
    if (idx < 8192 * 128) agg[idx] = 0.0f;
}

__global__ void residual_kernel(float* __restrict__ h,
                                const float* __restrict__ agg)
{
    int idx = blockIdx.x * blockDim.x + threadIdx.x;   // 8192*128 exact
    h[idx] = silu_f(h[idx] + agg[idx]);
}

__global__ void f2h_kernel(const float* __restrict__ in,
                           _Float16* __restrict__ outp, int n)
{
    int idx = blockIdx.x * blockDim.x + threadIdx.x;
    if (idx < n) outp[idx] = (_Float16)in[idx];
}

// ---------------------------------------------------------------------------
extern "C" void kernel_launch(void* const* d_in, const int* in_sizes, int n_in,
                              void* d_out, int out_size, void* d_ws, size_t ws_size,
                              hipStream_t stream)
{
    (void)in_sizes; (void)n_in; (void)out_size; (void)ws_size;

    const float* zero_dof_emb = (const float*)d_in[0];
    const float* inf_dof_emb  = (const float*)d_in[1];
    const float* inf_W        = (const float*)d_in[2];
    const float* inf_b        = (const float*)d_in[3];
    const float* dof_emb      = (const float*)d_in[4];
    const float* pos_emb      = (const float*)d_in[5];
    const float* sg_emb       = (const float*)d_in[6];
    const float* time_emb     = (const float*)d_in[7];
    // d_in[8..31]: gnn layer params (12 leaves per layer: aW0,ab0,aW1,ab1,aW2,ab2,
    //                                pW0,pb0,pW1,pb1,pW2,pb2)
    const float* zmW0 = (const float*)d_in[32];
    const float* zmb0 = (const float*)d_in[33];
    const float* zmW1 = (const float*)d_in[34];
    const float* zmb1 = (const float*)d_in[35];
    const float* zmW2 = (const float*)d_in[36];
    const float* zmb2 = (const float*)d_in[37];
    const float* imW0 = (const float*)d_in[38];
    const float* imb0 = (const float*)d_in[39];
    const float* imW1 = (const float*)d_in[40];
    const float* imb1 = (const float*)d_in[41];
    const float* imW2 = (const float*)d_in[42];
    const float* imb2 = (const float*)d_in[43];
    const int* edge_index = (const int*)d_in[44];
    const int* x_0_dof    = (const int*)d_in[45];
    const int* x_inf_dof  = (const int*)d_in[46];
    // d_in[47], d_in[48]: zero_idx / inf_idx == arange (implicit)
    const int* wyck       = (const int*)d_in[49];
    const int* sgI        = (const int*)d_in[50];
    const int* tI         = (const int*)d_in[51];
    float* out = (float*)d_out;

    // ---- workspace carve ----
    char* wp = (char*)d_ws;
    auto carve = [&](size_t bytes) -> void* {
        void* p = (void*)wp;
        wp += (bytes + 255) & ~(size_t)255;
        return p;
    };
    float*    h    = (float*)carve(8192ull * 128 * 4);
    float*    hdof = (float*)carve(8192ull * 64 * 4);
    _Float16* z    = (_Float16*)carve(8192ull * 192 * 2);
    float*    U    = (float*)carve(8192ull * 384 * 4);
    float*    Vb   = (float*)carve(8192ull * 384 * 4);
    _Float16* p1   = (_Float16*)carve(8192ull * 384 * 2);
    _Float16* p2   = (_Float16*)carve(8192ull * 384 * 2);
    float*    psi  = (float*)carve(8192ull * 128 * 4);
    float*    agg  = (float*)carve(8192ull * 128 * 4);
    _Float16* hz   = (_Float16*)carve(8192ull * 128 * 2);
    _Float16* t1   = (_Float16*)carve(4096ull * 256 * 2);
    _Float16* t2   = (_Float16*)carve(4096ull * 256 * 2);
    _Float16* Wa0h = (_Float16*)carve(147456ull * 2);
    _Float16* Wa1h = (_Float16*)carve(147456ull * 2);
    _Float16* Wp0h = (_Float16*)carve(73728ull * 2);
    _Float16* Wp1h = (_Float16*)carve(147456ull * 2);
    _Float16* Wp2h = (_Float16*)carve(49152ull * 2);
    _Float16* Wo0h = (_Float16*)carve(32768ull * 2);
    _Float16* Wo1h = (_Float16*)carve(65536ull * 2);
    _Float16* Wo2h = (_Float16*)carve(537600ull * 2);

    auto cvt = [&](const float* src, _Float16* dst, int n) {
        f2h_kernel<<<(n + 255) / 256, 256, 0, stream>>>(src, dst, n);
    };
    auto gemm = [&](const _Float16* A, const _Float16* W, const float* bias,
                    float* Cf, _Float16* Ch, int M, int K, int N, int doSilu) {
        dim3 grid((N + 63) / 64, M / 64);
        gemm_wmma_kernel<<<grid, 128, 0, stream>>>(A, W, bias, Cf, Ch, M, K, N, doSilu);
    };

    // ---- embeddings ----
    emb_h_kernel<<<8192, 128, 0, stream>>>(zero_dof_emb, inf_dof_emb, inf_W,
                                           inf_b, x_0_dof, x_inf_dof, h);
    hdof_kernel<<<(8192 * 64) / 256, 256, 0, stream>>>(dof_emb, pos_emb, sg_emb,
                                                       time_emb, wyck, sgI, tI, hdof);

    // ---- GNN layers ----
    for (int l = 0; l < 2; ++l) {
        const int base = 8 + 12 * l;
        const float* aW0 = (const float*)d_in[base + 0];
        const float* ab0 = (const float*)d_in[base + 1];
        const float* aW1 = (const float*)d_in[base + 2];
        const float* ab1 = (const float*)d_in[base + 3];
        const float* aW2 = (const float*)d_in[base + 4];
        const float* ab2 = (const float*)d_in[base + 5];
        const float* pW0 = (const float*)d_in[base + 6];
        const float* pb0 = (const float*)d_in[base + 7];
        const float* pW1 = (const float*)d_in[base + 8];
        const float* pb1 = (const float*)d_in[base + 9];
        const float* pW2 = (const float*)d_in[base + 10];
        const float* pb2 = (const float*)d_in[base + 11];

        cvt(aW0, Wa0h, 147456);
        cvt(aW1, Wa1h, 147456);
        cvt(pW0, Wp0h, 73728);
        cvt(pW1, Wp1h, 147456);
        cvt(pW2, Wp2h, 49152);

        buildz_kernel<<<(8192 * 192) / 256, 256, 0, stream>>>(h, hdof, z, agg);

        // Node-level precompute (a-MLP layer0 split + full psi-MLP)
        gemm(z, Wa0h,             ab0,     U,       nullptr, 8192, 192, 384, 0); // U = z·W0[:192]+b0
        gemm(z, Wa0h + 192 * 384, nullptr, Vb,      nullptr, 8192, 192, 384, 0); // V = z·W0[192:]
        gemm(z, Wp0h,             pb0,     nullptr, p1,      8192, 192, 384, 1);
        gemm(p1, Wp1h,            pb1,     nullptr, p2,      8192, 384, 384, 1);
        gemm(p2, Wp2h,            pb2,     psi,     nullptr, 8192, 384, 128, 0);

        // Edge-level attention + aggregation
        edge_kernel<<<1024, 256, 0, stream>>>(edge_index, U, Vb, ab1, Wa1h,
                                              aW2, ab2, psi, agg);

        residual_kernel<<<(8192 * 128) / 256, 256, 0, stream>>>(h, agg);
    }

    // ---- output heads ----
    cvt(h, hz, 8192 * 128);

    cvt(zmW0, Wo0h, 128 * 256);
    cvt(zmW1, Wo1h, 256 * 256);
    cvt(zmW2, Wo2h, 256 * 101);
    gemm(hz, Wo0h, zmb0, nullptr, t1, 4096, 128, 256, 1);
    gemm(t1, Wo1h, zmb1, nullptr, t2, 4096, 256, 256, 1);
    gemm(t2, Wo2h, zmb2, out, nullptr, 4096, 256, 101, 0);

    cvt(imW0, Wo0h, 128 * 256);
    cvt(imW1, Wo1h, 256 * 256);
    cvt(imW2, Wo2h, 256 * 2100);
    gemm(hz + 4096ull * 128, Wo0h, imb0, nullptr, t1, 4096, 128, 256, 1);
    gemm(t1, Wo1h, imb1, nullptr, t2, 4096, 256, 256, 1);
    gemm(t2, Wo2h, imb2, out + 4096ull * 101, nullptr, 4096, 256, 2100, 0);
}
